// Decoder_55216099557860
// MI455X (gfx1250) — matrix operand order repaired and checked by reading, not compiled, not run on hardware
//
#include <hip/hip_runtime.h>

// Problem constants (match reference setup_inputs)
#define B_  4
#define N_  4096
#define D_  1024
#define H_  16
#define FF_ 4096
#define M_  (B_ * N_)        // 16384 rows for the big GEMMs
#define DH_ 64               // head dim
#define TOT_ ((long)M_ * D_) // 16777216 elements per activation tensor

typedef __attribute__((ext_vector_type(16))) __bf16 bf16x16;
typedef __attribute__((ext_vector_type(8)))  float  f32x8;

union BFrag { uint4 q[2]; bf16x16 v; };   // 32 bytes = one WMMA 16-bit operand per lane

// Pack two f32 -> packed bf16 dword: round-half-up on bit15, then v_perm_b32
// grabs the two high halves in one instruction (3 VALU per pair).
__device__ __forceinline__ unsigned int pk2(float lo, float hi) {
  unsigned a = __float_as_uint(lo) + 0x8000u;
  unsigned b = __float_as_uint(hi) + 0x8000u;
  return __builtin_amdgcn_perm(b, a, 0x07060302u);
}
__device__ __forceinline__ uint4 pack8(const float* f) {
  uint4 u;
  u.x = pk2(f[0], f[1]); u.y = pk2(f[2], f[3]);
  u.z = pk2(f[4], f[5]); u.w = pk2(f[6], f[7]);
  return u;
}
__device__ __forceinline__ void ld8(const float* p, float* f) {
  float4 x0 = ((const float4*)p)[0], x1 = ((const float4*)p)[1];
  f[0] = x0.x; f[1] = x0.y; f[2] = x0.z; f[3] = x0.w;
  f[4] = x1.x; f[5] = x1.y; f[6] = x1.z; f[7] = x1.w;
}

// ---------------------------------------------------------------------------
// Generic bf16-WMMA GEMM:  C[M,N] = A[M,K] * Bw[N,K]^T  (+bias)(+resid)(+gelu)
// Block tile 128(M) x 64(N), 8 waves; wave w owns rows [w*16,w*16+16) x 64 cols.
// LDS holds tiles in WMMA *fragment order* ([tile][lane][16 bf16]) so operand
// loads are 2x ds_load_b128 and staging stores are 1x ds_store_b128 per 8 k.
// Double-buffered LDS, last k-step peeled: branch-free steady-state loop with
// one barrier per k-step; register prefetch + LDS stores overlap the WMMAs.
// A-frag lane map: lane<16 -> m=lane, k {0..7,16..23}; lane>=16 -> k {8..15,24..31}.
// B-frag: lane<16 -> n=lane, k {0..7,16..23}; lane>=16 -> k {8..15,24..31}.
// blockIdx.z batches (weights*V): zA/zC = (z/nH)*batch + (z%nH)*head, zB = z*bZ.
// EPI: 0=none, 1=+bias, 2=+bias+resid, 3=+bias+gelu(erf).
// ---------------------------------------------------------------------------
template <int EPI>
__global__ __launch_bounds__(256)
void gemm_bf16(const float* __restrict__ A, int lda, long aB, long aH, int nH,
               const float* __restrict__ Bw, int ldb, long bZ,
               const float* __restrict__ bias,
               const float* __restrict__ R, long cB, long cH,
               float* __restrict__ C, int ldc, int K) {
  __shared__ unsigned int As[2][8 * 32 * 8];   // 2 x 8KB
  __shared__ unsigned int Bs[2][4 * 32 * 8];   // 2 x 4KB

  const int tid  = threadIdx.x;
  const int wave = tid >> 5;
  const int lane = tid & 31;
  const int ml   = lane & 15;
  const int hi   = lane >> 4;

  const int z  = blockIdx.z;
  const long zA = (long)(z / nH) * aB + (long)(z % nH) * aH;
  const long zC = (long)(z / nH) * cB + (long)(z % nH) * cH;
  const float* Ag = A + zA;
  const float* Bg = Bw + (long)z * bZ;

  const int mBlk = blockIdx.y * 128;
  const int nBlk = blockIdx.x * 64;

  // Staging group decode (once): row gr = tid>>2, k-chunk gc = (tid&3)*8.
  const int gr = tid >> 2, gc = (tid & 3) << 3;
  const float* aP0 = Ag + (long)(mBlk + gr) * lda + gc;        // A rows 0..63
  const float* aP1 = Ag + (long)(mBlk + 64 + gr) * lda + gc;   // A rows 64..127
  const float* bP  = Bg + (long)(nBlk + gr) * ldb + gc;        // B rows 0..63

  // LDS fragment-order dword indices (loop invariant).
  const int ai0 = (((gr & 15) + (((gc >> 3) & 1) << 4)) + (gr >> 4) * 32) * 8 + (gc >> 4) * 4;
  const int ai1 = ai0 + 4 * 32 * 8;                             // rows 64..127 -> m-tiles 4..7
  const int bi  = (((gr & 15) + (((gc >> 3) & 1) << 4)) + (gr >> 4) * 32) * 8 + (gc >> 4) * 4;
  const int aOff = (wave * 32 + lane) * 8;

  float fA0[8], fA1[8], fB0[8];
  ld8(aP0, fA0); ld8(aP1, fA1); ld8(bP, fB0);
  *(uint4*)&As[0][ai0] = pack8(fA0);
  *(uint4*)&As[0][ai1] = pack8(fA1);
  *(uint4*)&Bs[0][bi]  = pack8(fB0);
  __syncthreads();

  f32x8 acc[4] = {};
  auto wmma_step = [&](int c) {
    BFrag a;
    const uint4* pa = (const uint4*)&As[c][aOff];
    a.q[0] = pa[0]; a.q[1] = pa[1];
#pragma unroll
    for (int nt = 0; nt < 4; ++nt) {
      BFrag b;
      const uint4* pb = (const uint4*)&Bs[c][(nt * 32 + lane) * 8];
      b.q[0] = pb[0]; b.q[1] = pb[1];
      acc[nt] = __builtin_amdgcn_wmma_f32_16x16x32_bf16(
          false, a.v, false, b.v, (short)0, acc[nt], false, false);
    }
  };

  int cur = 0;
  for (int k0 = 32; k0 < K; k0 += 32, cur ^= 1) {   // steady state (branch-free body)
    aP0 += 32; aP1 += 32; bP += 32;
    ld8(aP0, fA0); ld8(aP1, fA1); ld8(bP, fB0);     // prefetch next tile
    wmma_step(cur);                                  // compute current tile
    *(uint4*)&As[cur ^ 1][ai0] = pack8(fA0);         // stores overlap WMMAs
    *(uint4*)&As[cur ^ 1][ai1] = pack8(fA1);
    *(uint4*)&Bs[cur ^ 1][bi]  = pack8(fB0);
    __syncthreads();
  }
  wmma_step(cur);                                    // peeled last k-step

  // Epilogue. C 16x16 layout: VGPR r -> M = r + (lane>=16)*8, N = lane&15.
#pragma unroll
  for (int nt = 0; nt < 4; ++nt) {
    int col = nBlk + nt * 16 + ml;
    float bv = (EPI >= 1) ? bias[col] : 0.f;
#pragma unroll
    for (int r = 0; r < 8; ++r) {
      int row = mBlk + wave * 16 + r + (hi << 3);
      float val = acc[nt][r] + bv;
      if (EPI == 3) val = 0.5f * val * (1.f + erff(val * 0.70710678118654752f));
      long idx = zC + (long)row * ldc + col;
      if (EPI == 2) val += R[idx];
      C[idx] = val;
    }
  }
}

// ---------------------------------------------------------------------------
// S[b,h] = Q_blk^T * K_blk  (64x64, reduction dim = N_=4096), Q/K in [B,N,D].
// The transpose (A[m,k]=Q[k,m]) is absorbed into the staging *store* address:
// each thread reads an 8-long column (stride D_) and emits one ds_store_b128
// in fragment order, so compute-side operand loads are 2x ds_load_b128.
// One block per (b,h); 8 waves x 2 of the 16 (16x16) output tiles.
// ---------------------------------------------------------------------------
__global__ __launch_bounds__(256)
void score_kernel(const float* __restrict__ q, const float* __restrict__ k,
                  float* __restrict__ S) {
  __shared__ unsigned int Qs[2][4 * 32 * 8];   // 2 x 4KB, fragment order
  __shared__ unsigned int Ks[2][4 * 32 * 8];
  const int bh = blockIdx.x;
  const int b  = bh >> 4, h = bh & 15;
  const float* Qg = q + (long)b * N_ * D_ + h * DH_;
  const float* Kg = k + (long)b * N_ * D_ + h * DH_;

  const int tid = threadIdx.x, wave = tid >> 5, lane = tid & 31;
  const int ml = lane & 15, hi = lane >> 4;

  // Group decode once: column gi = tid>>2, n-chunk gn = (tid&3)*8.
  const int gi = tid >> 2, gn = (tid & 3) << 3;
  const float* qP = Qg + (long)gn * D_ + gi;
  const float* kP = Kg + (long)gn * D_ + gi;
  const int qi = (((gi & 15) + (((gn >> 3) & 1) << 4)) + (gi >> 4) * 32) * 8 + (gn >> 4) * 4;
  const int ki = (((gi & 15) + (((gn >> 3) & 1) << 4)) + (gi >> 4) * 32) * 8 + (gn >> 4) * 4;

  auto ldCol = [&](const float* p, float* f) {
#pragma unroll
    for (int t = 0; t < 8; ++t) f[t] = p[(long)t * D_];
  };

  float fq[8], fk[8];
  ldCol(qP, fq); ldCol(kP, fk);
  *(uint4*)&Qs[0][qi] = pack8(fq);
  *(uint4*)&Ks[0][ki] = pack8(fk);
  __syncthreads();

  const int t0 = wave * 2, tiQ = t0 >> 2;
  const int aOff = (tiQ * 32 + lane) * 8;
  f32x8 acc[2] = {};
  auto wmma_step = [&](int c) {
    BFrag a;
    const uint4* pa = (const uint4*)&Qs[c][aOff];
    a.q[0] = pa[0]; a.q[1] = pa[1];
#pragma unroll
    for (int tt = 0; tt < 2; ++tt) {
      int tj = (t0 + tt) & 3;
      BFrag bf;
      const uint4* pb = (const uint4*)&Ks[c][(tj * 32 + lane) * 8];
      bf.q[0] = pb[0]; bf.q[1] = pb[1];
      acc[tt] = __builtin_amdgcn_wmma_f32_16x16x32_bf16(
          false, a.v, false, bf.v, (short)0, acc[tt], false, false);
    }
  };

  int cur = 0;
  for (int n0 = 32; n0 < N_; n0 += 32, cur ^= 1) {   // steady state
    qP += (long)32 * D_; kP += (long)32 * D_;
    ldCol(qP, fq); ldCol(kP, fk);
    wmma_step(cur);
    *(uint4*)&Qs[cur ^ 1][qi] = pack8(fq);
    *(uint4*)&Ks[cur ^ 1][ki] = pack8(fk);
    __syncthreads();
  }
  wmma_step(cur);                                     // peeled last n-step

#pragma unroll
  for (int tt = 0; tt < 2; ++tt) {
    int t = t0 + tt, ti = t >> 2, tj = t & 3;
#pragma unroll
    for (int r = 0; r < 8; ++r) {
      int i = ti * 16 + r + (hi << 3);
      int j = tj * 16 + ml;
      S[(long)bh * 4096 + i * 64 + j] = acc[tt][r];
    }
  }
}

// weights[bh,i,j] = | sum_m S[bh,i,m] * exp(-2*pi*i*j*m/64) |
__global__ __launch_bounds__(256)
void weights_kernel(const float* __restrict__ S, float* __restrict__ W) {
  __shared__ float ct[64], st[64];
  if (threadIdx.x < 64) {
    float ang = -0.0981747704246810387f * (float)threadIdx.x;  // -2*pi/64 * t
    __sincosf(ang, &st[threadIdx.x], &ct[threadIdx.x]);
  }
  __syncthreads();
  int idx = blockIdx.x * 256 + threadIdx.x;   // B_*H_*64*64 = 262144 total
  int j = idx & 63, i = (idx >> 6) & 63, bh = idx >> 12;
  const float* row = S + (long)bh * 4096 + i * 64;
  float re = 0.f, im = 0.f;
#pragma unroll 8
  for (int m = 0; m < 64; ++m) {
    int t = (j * m) & 63;
    float v = row[m];
    re += v * ct[t];
    im += v * st[t];
  }
  W[idx] = sqrtf(re * re + im * im);
}

// Global-std reduction helpers (ddof=1 over the whole tensor)
__global__ void zero_stats(float* s) { s[0] = 0.f; s[1] = 0.f; }

__global__ __launch_bounds__(256)
void reduce_kernel(const float* __restrict__ x, long n, float* __restrict__ stats) {
  __shared__ float ss[256], sq[256];
  float s = 0.f, q = 0.f;
  for (long i = (long)blockIdx.x * 256 + threadIdx.x; i < n; i += (long)gridDim.x * 256) {
    float v = x[i];
    s += v; q += v * v;
  }
  ss[threadIdx.x] = s; sq[threadIdx.x] = q;
  __syncthreads();
  for (int o = 128; o > 0; o >>= 1) {
    if (threadIdx.x < o) { ss[threadIdx.x] += ss[threadIdx.x + o]; sq[threadIdx.x] += sq[threadIdx.x + o]; }
    __syncthreads();
  }
  if (threadIdx.x == 0) { atomicAdd(&stats[0], ss[0]); atomicAdd(&stats[1], sq[0]); }
}

__global__ __launch_bounds__(256)
void norm_kernel(const float* __restrict__ x, const float* __restrict__ stats,
                 const float* __restrict__ alpha, const float* __restrict__ beta,
                 const float* __restrict__ eps, float* __restrict__ out, long n) {
  long i = (long)blockIdx.x * 256 + threadIdx.x;
  if (i >= n) return;
  float cnt = (float)n;
  float mean = stats[0] / cnt;
  float var = (stats[1] - cnt * mean * mean) / (cnt - 1.f);
  float sd = sqrtf(fmaxf(var, 0.f));
  int d = (int)(i & (D_ - 1));
  out[i] = x[i] / (sd + eps[d]) * alpha[d] + beta[d];
}

// ---------------------------------------------------------------------------
extern "C" void kernel_launch(void* const* d_in, const int* in_sizes, int n_in,
                              void* d_out, int out_size, void* d_ws, size_t ws_size,
                              hipStream_t stream) {
  const float* x      = (const float*)d_in[0];
  const float* wq_w   = (const float*)d_in[1];
  const float* wq_b   = (const float*)d_in[2];
  const float* wk_w   = (const float*)d_in[3];
  const float* wk_b   = (const float*)d_in[4];
  const float* wv_w   = (const float*)d_in[5];
  const float* wv_b   = (const float*)d_in[6];
  const float* wo_w   = (const float*)d_in[7];
  const float* wo_b   = (const float*)d_in[8];
  const float* mlp_w1 = (const float*)d_in[9];
  const float* mlp_b1 = (const float*)d_in[10];
  const float* mlp_w2 = (const float*)d_in[11];
  const float* mlp_b2 = (const float*)d_in[12];
  const float* an_a   = (const float*)d_in[13];
  const float* an_b   = (const float*)d_in[14];
  const float* an_e   = (const float*)d_in[15];
  const float* mn_a   = (const float*)d_in[16];
  const float* mn_b   = (const float*)d_in[17];
  const float* mn_e   = (const float*)d_in[18];
  float* out = (float*)d_out;

  // Workspace layout (floats). Buffers reused once dead to bound footprint.
  float* ws    = (float*)d_ws;
  float* stats = ws;                     // 64
  float* q     = stats + 64;             // TOT_
  float* k     = q + TOT_;               // TOT_
  float* v     = k + TOT_;               // TOT_
  float* hbuf  = v + TOT_;               // M_*FF_ = 67108864
  float* S     = hbuf + (long)M_ * FF_;  // 262144
  float* W     = S + 262144;             // 262144
  float* attn_pre = q;   // q dead after score
  float* y        = k;   // k dead after score
  float* attn_n   = v;   // v dead after weights*V
  float* out2     = q;   // attn_pre dead after Wo

  const dim3 blk(256);
  const dim3 gBig(D_ / 64, M_ / 128, 1);    // N=1024 outputs
  const dim3 gFF (FF_ / 64, M_ / 128, 1);   // N=4096 outputs

  // 1) QKV projections (bias)
  gemm_bf16<1><<<gBig, blk, 0, stream>>>(x, D_, 0, 0, 1, wq_w, D_, 0, wq_b, nullptr, 0, 0, q, D_, D_);
  gemm_bf16<1><<<gBig, blk, 0, stream>>>(x, D_, 0, 0, 1, wk_w, D_, 0, wk_b, nullptr, 0, 0, k, D_, D_);
  gemm_bf16<1><<<gBig, blk, 0, stream>>>(x, D_, 0, 0, 1, wv_w, D_, 0, wv_b, nullptr, 0, 0, v, D_, D_);

  // 2) S = Q^T K per (b,h); 3) weights = |DFT64(rows of S)|
  score_kernel<<<dim3(B_ * H_), blk, 0, stream>>>(q, k, S);
  weights_kernel<<<dim3(262144 / 256), blk, 0, stream>>>(S, W);

  // 4) attn_pre[b,n,h*64+i] = sum_j W[bh,i,j] * v[b,n,h*64+j]   (batched over b,h)
  gemm_bf16<0><<<dim3(1, N_ / 128, B_ * H_), blk, 0, stream>>>(
      v, D_, (long)N_ * D_, DH_, H_, W, DH_, 4096, nullptr, nullptr,
      (long)N_ * D_, DH_, attn_pre, D_, DH_);

  // 5) Wo projection + residual(x)
  gemm_bf16<2><<<gBig, blk, 0, stream>>>(attn_pre, D_, 0, 0, 1, wo_w, D_, 0, wo_b, x, 0, 0, y, D_, D_);

  // 6) global-std norm #1
  zero_stats<<<dim3(1), dim3(1), 0, stream>>>(stats);
  reduce_kernel<<<dim3(1024), blk, 0, stream>>>(y, TOT_, stats);
  norm_kernel<<<dim3(TOT_ / 256), blk, 0, stream>>>(y, stats, an_a, an_b, an_e, attn_n, TOT_);

  // 7) MLP: gelu(attn_n @ w1^T + b1) @ w2^T + b2 + attn_n
  gemm_bf16<3><<<gFF, blk, 0, stream>>>(attn_n, D_, 0, 0, 1, mlp_w1, D_, 0, mlp_b1, nullptr, 0, 0, hbuf, FF_, D_);
  gemm_bf16<2><<<gBig, blk, 0, stream>>>(hbuf, FF_, 0, 0, 1, mlp_w2, FF_, 0, mlp_b2, attn_n, 0, 0, out2, D_, FF_);

  // 8) global-std norm #2 -> output
  zero_stats<<<dim3(1), dim3(1), 0, stream>>>(stats);
  reduce_kernel<<<dim3(1024), blk, 0, stream>>>(out2, TOT_, stats);
  norm_kernel<<<dim3(TOT_ / 256), blk, 0, stream>>>(out2, stats, mn_a, mn_b, mn_e, out, TOT_);
}
